// S5Block_60361470378298
// MI455X (gfx1250) — matrix operand order repaired and checked by reading, not compiled, not run on hardware
//
#include <hip/hip_runtime.h>
#include <hip/hip_bf16.h>

#define DM   96
#define DI   192
#define NST  16
#define RNK  6
#define KD   4
#define BB   8
#define HH   64
#define WW   64
#define LL   (HH*WW)
#define NTOK (BB*LL)          // 32768 tokens
#define CPROJ 38              // RNK + 2*NST
#define NXP  (KD*CPROJ)       // 152 stacked x_proj outputs

typedef __bf16 bf16_t;
typedef bf16_t       v16bf __attribute__((ext_vector_type(16)));
typedef float        v8f   __attribute__((ext_vector_type(8)));
typedef unsigned int u32x4 __attribute__((ext_vector_type(4)));
typedef int          i32x8 __attribute__((ext_vector_type(8)));
typedef int          i32x4 __attribute__((ext_vector_type(4)));

__device__ __forceinline__ bf16_t tobf(float f) { return (bf16_t)f; }

// ---------------------------------------------------------------------------
// TDM: DMA a 16 x KDIM f32 row strip of A (row-major, lda=KDIM) into LDS.
// D# per CDNA5 ISA 8.3/8.4: 2D tensor, data_size=4B, tensor_dim0=tile_dim0=
// KDIM, tensor_dim1=tile_dim1=16 rows, dim0_stride=KDIM, lds_addr=0 (the
// tile is the kernel's only LDS allocation). Issued by wave 0 only; groups
// are workgroup-uniform so they live in SGPRs. Tracked on TENSORcnt.
// This toolchain exposes the 6-arg builtin (g0, g1, g2, g3, g4, cpol).
// ---------------------------------------------------------------------------
template <int KDIM>
__device__ __forceinline__ void tdm_load_strip(const float* gA)
{
    unsigned long long ga = (unsigned long long)gA;
    u32x4 g0 = { 1u,                                   // count=1, load, no gather
                 0u,                                   // lds_addr = 0
                 (unsigned)(ga & 0xFFFFFFFFull),       // global_addr[31:0]
                 (unsigned)((ga >> 32) & 0x1FFFFFFull) // global_addr[56:32]
                     | (2u << 30) };                   // type = 2 (image)
    i32x8 g1 = { (int)(2u << 16),                      // data_size = 2 (4 bytes)
                 (int)((KDIM & 0xFFFF) << 16),         // tensor_dim0 lo
                 (int)(16u << 16),                     // tensor_dim0 hi=0 | tensor_dim1=16
                 (int)((KDIM & 0xFFFF) << 16),         // tensor_dim1 hi=0 | tile_dim0=KDIM
                 16,                                   // tile_dim1=16, tile_dim2=0
                 KDIM,                                 // tensor_dim0_stride lo
                 0, 0 };                               // stride hi / dim1_stride unused (2D)
    i32x4 gz4 = { 0, 0, 0, 0 };
    i32x8 gz8 = { 0, 0, 0, 0, 0, 0, 0, 0 };
    __builtin_amdgcn_tensor_load_to_lds(g0, g1, gz4, gz4, gz8, 0);
    __builtin_amdgcn_s_wait_tensorcnt(0);
}

// ---------------------------------------------------------------------------
// Workgroup GEMM: block = 256 threads = 8 waves; each block owns a 16-row M
// strip of C = A * B.  A strip staged once into LDS by the TDM, then every
// wave walks its share of the NTILES 16-wide N tiles on the matrix pipe with
// v_wmma_f32_16x16x32_bf16 (bf16 in, f32 accumulate).  B is an element
// getter bget(k,n) (weights are KB-scale, L2-resident).
// Fragment packing per ISA 7.12.2 (wave32):
//   A 16x32 : lane[3:0]=M row, lane[4]=K half; elems 0..7 = K half*8.., 8..15 = K 16+half*8..
//   B 32x16 : lane[3:0]=N col, lane[4]=K half; elems 0..7 = K half*8.., 8..15 = K 16+half*8..
//   D 16x16 : elem i -> M = i + 8*lane[4], N = lane[3:0]
// The asm-escape after the barrier tells the compiler the LDS tile was
// written (by the TDM) so the ds_load path is kept and the allocation lives.
// ---------------------------------------------------------------------------
template <int KDIM, int NTILES, typename BGet>
__device__ __forceinline__ void gemm_rowstrip(const float* __restrict__ A,
                                              BGet bget, float* __restrict__ C,
                                              int ldc, int Ncols,
                                              float* lds)
{
    const int m0   = blockIdx.x * 16;
    const int wave = threadIdx.x >> 5;
    if (wave == 0)
        tdm_load_strip<KDIM>(A + (size_t)m0 * KDIM);
    __syncthreads();                                   // LDS strip visible to all waves
    asm volatile("" : : "v"(lds) : "memory");          // TDM wrote LDS: keep alloc, force reloads

    const int lane = threadIdx.x & 31;
    const int mn   = lane & 15;
    const int half = lane >> 4;
    for (int nt = wave; nt < NTILES; nt += 8) {
        const int  n0  = nt * 16;
        const int  n   = n0 + mn;
        const bool nok = (n < Ncols);
        v8f acc = {};
        for (int k0 = 0; k0 < KDIM; k0 += 32) {
            v16bf a, b;
            const float* arow = lds + mn * KDIM + k0;  // ds_load_b128 path
#pragma unroll
            for (int e = 0; e < 8; ++e) {
                a[e]     = tobf(arow[half * 8 + e]);
                a[e + 8] = tobf(arow[16 + half * 8 + e]);
            }
#pragma unroll
            for (int e = 0; e < 16; ++e)
                b[e] = tobf(nok ? bget(k0 + (e & 8) * 2 + half * 8 + (e & 7), n) : 0.f);
            acc = __builtin_amdgcn_wmma_f32_16x16x32_bf16(
                      false, a, false, b, (short)0, acc, false, false);
        }
        if (nok) {
#pragma unroll
            for (int i = 0; i < 8; ++i)
                C[(size_t)(m0 + half * 8 + i) * ldc + n] = acc[i];
        }
    }
}

// ---------------- Stage 1: xz = x @ in_proj_w^T   (32768x96)*(96x384) -------
__global__ void k_in_proj(const float* __restrict__ X,
                          const float* __restrict__ Wt,   // (384,96) row-major
                          float* __restrict__ XZ)
{
    __shared__ float tileA[16 * DM];
    auto bget = [&](int k, int n) { return Wt[n * DM + k]; };
    gemm_rowstrip<DM, 24>(X, bget, XZ, 2 * DI, 2 * DI, tileA);
}

// ---------------- Stage 2: depthwise 3x3 conv (SAME) + SiLU -----------------
__global__ void k_conv(const float* __restrict__ XZ,
                       const float* __restrict__ CW,     // (192,1,3,3)
                       const float* __restrict__ CB,
                       float* __restrict__ XC)           // (token,192)
{
    int idx = blockIdx.x * blockDim.x + threadIdx.x;
    if (idx >= NTOK * DI) return;
    int d = idx % DI, tok = idx / DI;
    int b = tok / LL, l = tok % LL, h = l / WW, w = l % WW;
    float s = CB[d];
#pragma unroll
    for (int dy = -1; dy <= 1; ++dy) {
        int hy = h + dy;
        if ((unsigned)hy >= HH) continue;
#pragma unroll
        for (int dx = -1; dx <= 1; ++dx) {
            int wx = w + dx;
            if ((unsigned)wx >= WW) continue;
            s += XZ[(size_t)(b * LL + hy * WW + wx) * (2 * DI) + d] *
                 CW[d * 9 + (dy + 1) * 3 + (dx + 1)];
        }
    }
    float sig = 1.f / (1.f + __expf(-s));
    XC[(size_t)tok * DI + d] = s * sig;                  // SiLU
}

// ---------------- Stage 3: fused 4-direction x_proj -------------------------
// x_dbl[token, k*38+c] = sum_d xc[token,d] * x_proj_weight[k,c,d]
__global__ void k_xproj(const float* __restrict__ XC,
                        const float* __restrict__ XPW,   // (4,38,192)
                        float* __restrict__ XDBL)        // (token,152)
{
    __shared__ float tileA[16 * DI];
    auto bget = [&](int k, int n) {
        int kk = n / CPROJ, c = n % CPROJ;
        return XPW[(kk * CPROJ + c) * DI + k];
    };
    gemm_rowstrip<DI, 10>(XC, bget, XDBL, NXP, NXP, tileA);
}

// ---------------- Stage 4: selective scan -----------------------------------
// One block per (b,k); thread d keeps its 16 states in VGPRs for 4096 steps.
// Per-step dt/B/C row address depends only on blockIdx+t => scalar loads.
__global__ void k_scan(const float* __restrict__ XC,
                       const float* __restrict__ XDBL,
                       const float* __restrict__ DTW,    // (4,192,6)
                       const float* __restrict__ DTB,    // (4,192)
                       const float* __restrict__ ALOG,   // (4*192,16)
                       const float* __restrict__ DSv,    // (4*192)
                       float* __restrict__ YB)           // (b,k,lhw,192)
{
    const int b = blockIdx.x >> 2;
    const int k = blockIdx.x & 3;
    const int d = threadIdx.x;                           // 0..191
    float A[NST], h[NST], wdt[RNK];
#pragma unroll
    for (int n = 0; n < NST; ++n) {
        A[n] = -__expf(ALOG[(k * DI + d) * NST + n]);
        h[n] = 0.f;
    }
#pragma unroll
    for (int r = 0; r < RNK; ++r) wdt[r] = DTW[(k * DI + d) * RNK + r];
    const float bias = DTB[k * DI + d];
    const float Dval = DSv[k * DI + d];

    for (int t = 0; t < LL; ++t) {
        int t2  = (k >= 2) ? (LL - 1 - t) : t;
        int lhw = (k & 1) ? ((t2 % HH) * WW + (t2 / HH)) : t2;  // wh vs hw order
        const float* row = XDBL + (size_t)(b * LL + lhw) * NXP + k * CPROJ;
        float u = XC[(size_t)(b * LL + lhw) * DI + d];
        float s = bias;
#pragma unroll
        for (int r = 0; r < RNK; ++r) s += wdt[r] * row[r];     // dt projection
        float delta = (s > 20.f) ? s : __logf(1.f + __expf(s)); // softplus
        float du = delta * u;
        float y  = 0.f;
#pragma unroll
        for (int n = 0; n < NST; ++n) {
            float dA = __expf(delta * A[n]);
            h[n] = dA * h[n] + du * row[RNK + n];               // B_t
            y   += h[n] * row[RNK + NST + n];                   // C_t
        }
        YB[(size_t)((b * KD + k) * LL + lhw) * DI + d] = y + Dval * u;
    }
}

// ---------------- Stage 5: combine dirs + LayerNorm + SiLU gate -------------
__global__ void k_ln(const float* __restrict__ YB,
                     const float* __restrict__ XZ,       // z = channels 192..383
                     const float* __restrict__ LW,
                     const float* __restrict__ LB,
                     float* __restrict__ YN)             // (token,192)
{
    int wv   = blockIdx.x * (blockDim.x >> 5) + (threadIdx.x >> 5);
    int lane = threadIdx.x & 31;
    if (wv >= NTOK) return;
    int b = wv / LL, l = wv % LL;
    float v[6], s = 0.f, s2 = 0.f;
#pragma unroll
    for (int j = 0; j < 6; ++j) {
        int d = j * 32 + lane;
        float y = 0.f;
#pragma unroll
        for (int k = 0; k < KD; ++k)
            y += YB[(size_t)((b * KD + k) * LL + l) * DI + d];
        v[j] = y; s += y; s2 += y * y;
    }
#pragma unroll
    for (int off = 16; off; off >>= 1) {
        s  += __shfl_xor(s,  off, 32);
        s2 += __shfl_xor(s2, off, 32);
    }
    float mu  = s / DI;
    float var = s2 / DI - mu * mu;
    float inv = rsqrtf(var + 1e-5f);
#pragma unroll
    for (int j = 0; j < 6; ++j) {
        int d   = j * 32 + lane;
        float z = XZ[(size_t)wv * (2 * DI) + DI + d];
        float sz = z / (1.f + __expf(-z));               // SiLU(z)
        YN[(size_t)wv * DI + d] = ((v[j] - mu) * inv * LW[d] + LB[d]) * sz;
    }
}

// ---------------- Stage 6: out = yn @ out_proj_w^T  (32768x192)*(192x96) ----
__global__ void k_out_proj(const float* __restrict__ YN,
                           const float* __restrict__ OW, // (96,192) row-major
                           float* __restrict__ OUT)
{
    __shared__ float tileA[16 * DI];
    auto bget = [&](int k, int n) { return OW[n * DI + k]; };
    gemm_rowstrip<DI, 6>(YN, bget, OUT, DM, DM, tileA);
}

extern "C" void kernel_launch(void* const* d_in, const int* in_sizes, int n_in,
                              void* d_out, int out_size, void* d_ws, size_t ws_size,
                              hipStream_t stream)
{
    const float* x    = (const float*)d_in[0];
    const float* ipw  = (const float*)d_in[1];
    const float* cw   = (const float*)d_in[2];
    const float* cb   = (const float*)d_in[3];
    const float* xpw  = (const float*)d_in[4];
    const float* dtw  = (const float*)d_in[5];
    const float* dtb  = (const float*)d_in[6];
    const float* alog = (const float*)d_in[7];
    const float* dsv  = (const float*)d_in[8];
    const float* lnw  = (const float*)d_in[9];
    const float* lnb  = (const float*)d_in[10];
    const float* opw  = (const float*)d_in[11];

    float* ws   = (float*)d_ws;
    float* XZ   = ws;                                    // 32768*384
    float* XC   = XZ   + (size_t)NTOK * 2 * DI;          // 32768*192
    float* XDBL = XC   + (size_t)NTOK * DI;              // 32768*152
    float* YB   = XDBL + (size_t)NTOK * NXP;             // 4*32768*192
    float* YN   = YB   + (size_t)NTOK * KD * DI;         // 32768*192

    // GEMMs: one 16-row M strip per 256-thread block (2048 blocks).
    k_in_proj<<<NTOK / 16, 256, 0, stream>>>(x, ipw, XZ);
    k_conv<<<(NTOK * DI + 255) / 256, 256, 0, stream>>>(XZ, cw, cb, XC);
    k_xproj<<<NTOK / 16, 256, 0, stream>>>(XC, xpw, XDBL);
    k_scan<<<BB * KD, DI, 0, stream>>>(XC, XDBL, dtw, dtb, alog, dsv, YB);
    k_ln<<<NTOK / 8, 256, 0, stream>>>(YB, XZ, lnw, lnb, YN);
    k_out_proj<<<NTOK / 16, 256, 0, stream>>>(YN, opw, (float*)d_out);
}
